// MultiGraphAttentionLayer_37366215475451
// MI455X (gfx1250) — matrix operand order repaired and checked by reading, not compiled, not run on hardware
//
#include <hip/hip_runtime.h>
#include <cstdint>
#include <cstddef>

// ---------------------------------------------------------------------------
// MI455X (gfx1250) GAT forward — all three GEMM families on WMMA f16->f32:
//   (1) xW = outputs @ W^T + b        (k_xw_wmma)
//   (2) AxW = softmax(att) @ xW       (k_att, attention generated in-register)
//   (3) out = h @ Wout^T + bout       (k_out_wmma)
// Activations move between stages as f16 WMMA fragments (L2-resident).
// ---------------------------------------------------------------------------

typedef __attribute__((ext_vector_type(16))) _Float16 v16h;
typedef __attribute__((ext_vector_type(8)))  float    v8f;
typedef _Float16 f16;

#define HEADS 3
#define D     150
#define DP    160          // D padded to 10 WMMA tiles
#define IN    300
#define INP   320          // IN padded to 10 K-chunks of 32
#define F1    450
#define BB    8
#define N     1024
#define BN    (BB * N)
#define RT    (BN / 16)    // 512 row tiles
#define ALPHA 0.2f
#define NEGC  (-9.0e15f)

// Fragment layout conventions for v_wmma_f32_16x16x32_f16 (wave32, ISA 7.12.2):
//  A (16x32): lane = M&15 (+16 selects K+8 group), element e: K = e + 8*((e>=8)+(lane>=16))
//  B (32x16): lane = N&15 (+16 selects K+8 group), element e: K = e + 8*((e>=8)+(lane>=16))
//  C/D:       VGPR r, lane L -> M = r + 8*(L>=16), N = L&15

#define WMMA(av, bv, c) \
  __builtin_amdgcn_wmma_f32_16x16x32_f16(false, (av), false, (bv), (short)0, (c), false, false)

// Inverse A-layout map: column index kk (0..31) -> (laneHi, e)
__device__ __forceinline__ void a_slot(int kk, int& laneHi, int& e) {
  laneHi = (kk >> 3) & 1;
  e = (kk & 7) + ((kk >= 16) ? 8 : 0);
}

// ---------------------------------------------------------------------------
// zero helper (pad columns of the h fragment buffer must be 0 every call)
// ---------------------------------------------------------------------------
__global__ void k_zero_u32(uint32_t* __restrict__ p, int n) {
  int i = blockIdx.x * blockDim.x + threadIdx.x;
  if (i < n) p[i] = 0u;
}

// ---------------------------------------------------------------------------
// k_prep_x: x (f32, [BN][IN]) -> A-fragment f16 buffer [RT][10][32][16]
// ---------------------------------------------------------------------------
__global__ void __launch_bounds__(INP) k_prep_x(const float* __restrict__ x,
                                                f16* __restrict__ xA) {
  const int row = blockIdx.x;
  const int f   = threadIdx.x;                 // 0..319
  const float v = (f < IN) ? x[(size_t)row * IN + f] : 0.f;
  const int rowTile = row >> 4, M = row & 15;
  const int kc = f >> 5, kk = f & 31;
  int laneHi, e; a_slot(kk, laneHi, e);
  xA[(((size_t)rowTile * 10 + kc) * 32 + (M + 16 * laneHi)) * 16 + e] = (f16)v;
}

// ---------------------------------------------------------------------------
// k_prep_w: weight matrix -> B-fragment f16 buffer [kcCount][dtCount][32][16]
//   B[k=f', n=d] = W[d*ldW + colOff + fmap(f')]
//   fmap: f'<320 -> f' (valid if < fc0); f'>=320 -> fc0 + (f'-320) (valid if < Ftot)
//   (handles the x|AxW concat gap for layer 1: 450 real -> 480 padded)
// ---------------------------------------------------------------------------
__global__ void k_prep_w(const float* __restrict__ W, int ldW, int colOff,
                         int Dr, int dtCount, int kcCount, int fc0, int Ftot,
                         f16* __restrict__ frag) {
  const int idx = blockIdx.x * blockDim.x + threadIdx.x;
  const int total = kcCount * dtCount * 512;
  if (idx >= total) return;
  const int e    = idx & 15;
  const int lane = (idx >> 4) & 31;
  const int rest = idx >> 9;
  const int dt   = rest % dtCount;
  const int kc   = rest / dtCount;
  const int d    = dt * 16 + (lane & 15);
  const int fp   = kc * 32 + ((lane < 16) ? 0 : 8) + (e & 7) + ((e >= 8) ? 16 : 0);   // B-layout K
  int f; bool valid;
  if (fp < INP) { f = fp;               valid = (f < fc0);  }
  else          { f = fc0 + (fp - INP); valid = (f < Ftot); }
  const float v = (valid && d < Dr) ? W[(size_t)d * ldW + colOff + f] : 0.f;
  frag[idx] = (f16)v;
}

// ---------------------------------------------------------------------------
// k_xw_wmma: one wave per 16-row tile.
//   acc[dt] += A(outputs chunk) x B(W chunk), dt = 0..9  (D padded 160)
//   A chunks 0..9 from xA (x), chunks 10..14 from axwA (AxW_prev, layer 1).
//   Epilogue: +bias, s1/s2 via shfl reduction, scatter xW into B-fragment
//   layout consumed by k_att.
// ---------------------------------------------------------------------------
__global__ void __launch_bounds__(128) k_xw_wmma(
    const f16*  __restrict__ xA,     // [RT][10][32][16]
    const f16*  __restrict__ axwA,   // [RT][5][32][16] (layer 1) or nullptr
    const f16*  __restrict__ wfrag,  // [kcCount][10][32][16]
    const float* __restrict__ bias,  // [D]
    const float* __restrict__ a1,    // [D]
    const float* __restrict__ a2,    // [D]
    int kcCount,                     // 10 (layer 0) or 15 (layer 1)
    f16*   __restrict__ xW16,        // [BB][32][10][32][16] B-layout
    float* __restrict__ s1,
    float* __restrict__ s2) {
  const int w = threadIdx.x >> 5;
  const int lane = threadIdx.x & 31;
  const int rowTile = blockIdx.x * 4 + w;      // 0..RT-1

  v8f acc[10];
  #pragma unroll
  for (int dt = 0; dt < 10; ++dt) {
    #pragma unroll
    for (int r = 0; r < 8; ++r) acc[dt][r] = 0.f;
  }

  for (int kc = 0; kc < kcCount; ++kc) {
    const f16* asrc = (kc < 10)
        ? xA   + (((size_t)rowTile * 10 + kc) * 32 + lane) * 16
        : axwA + (((size_t)rowTile * 5 + (kc - 10)) * 32 + lane) * 16;
    const v16h av = *reinterpret_cast<const v16h*>(asrc);
    const f16* wc = wfrag + (size_t)kc * (10 * 512);
    if (kc + 1 < kcCount) __builtin_prefetch(wc + 10 * 512, 0, 0);
    #pragma unroll
    for (int dt = 0; dt < 10; ++dt) {
      const v16h bv = *reinterpret_cast<const v16h*>(wc + ((size_t)(dt * 32 + lane) << 4));
      acc[dt] = WMMA(av, bv, acc[dt]);
    }
  }

  const int half = lane >> 4, laneLo = lane & 15;
  float p1[8], p2[8];
  #pragma unroll
  for (int r = 0; r < 8; ++r) { p1[r] = 0.f; p2[r] = 0.f; }

  #pragma unroll
  for (int dt = 0; dt < 10; ++dt) {
    const int d = dt * 16 + laneLo;
    const float bvl = (d < D) ? bias[d] : 0.f;
    const float a1v = (d < D) ? a1[d] : 0.f;
    const float a2v = (d < D) ? a2[d] : 0.f;
    #pragma unroll
    for (int r = 0; r < 8; ++r) {
      const int M = r + 8 * half;
      const float v = (d < D) ? (acc[dt][r] + bvl) : 0.f;
      p1[r] += v * a1v;
      p2[r] += v * a2v;
      // scatter into k_att's B-fragment layout
      const int gRow = rowTile * 16 + M;
      const int b = gRow >> 10, n = gRow & (N - 1);
      const int kcB = n >> 5, kk = n & 31;
      const int laneB = laneLo + ((kk & 8) ? 16 : 0);
      const int eB = (kk & 7) + ((kk & 16) ? 8 : 0);
      xW16[((((size_t)b * 32 + kcB) * 10 + dt) * 32 + laneB) * 16 + eB] = (f16)v;
    }
  }

  // reduce s1/s2 partials across the 16 lanes of each half (xor stays in-half)
  #pragma unroll
  for (int off = 8; off >= 1; off >>= 1) {
    #pragma unroll
    for (int r = 0; r < 8; ++r) {
      p1[r] += __shfl_xor(p1[r], off, 32);
      p2[r] += __shfl_xor(p2[r], off, 32);
    }
  }
  if (laneLo == 0) {
    #pragma unroll
    for (int r = 0; r < 8; ++r) {
      const int gRow = rowTile * 16 + r + 8 * half;
      s1[gRow] = p1[r];
      s2[gRow] = p2[r];
    }
  }
}

// ---------------------------------------------------------------------------
// k_att: AxW = relu( softmax(mask(leaky(s1[i]+s2[k]))) @ xW )
// One wave per 16-row tile; attention built in-register (rank-1 + mask).
// Epilogue writes (a) AxW in A-fragment layout (layer-1 input, if axwA!=null)
// and (b) AxW + x into the concatenated h A-fragment buffer at fBase.
// ---------------------------------------------------------------------------
__global__ void __launch_bounds__(128) k_att(
    const float* __restrict__ adj,
    const float* __restrict__ s1g,
    const float* __restrict__ s2g,
    const f16*   __restrict__ xW16,
    const float* __restrict__ x,
    f16* __restrict__ h16,           // [RT][30][32][16]
    f16* __restrict__ axwA,          // [RT][5][32][16] or nullptr
    int fBase,                       // j*320 + (layer ? 150 : 0)
    int xBase) {                     // layer ? 150 : 0
  __shared__ float smax[4][16];
  __shared__ float ssum[4][16];

  const int w    = threadIdx.x >> 5;
  const int lane = threadIdx.x & 31;
  const int tile = blockIdx.x * 4 + w;         // 0..511
  const int b    = tile >> 6;
  const int rowBase = (tile & 63) << 4;

  const float* adjB = adj + (size_t)b * N * N;
  const float* s2b  = s2g + b * N;

  // ---- pass 1: per-row online softmax stats ----
  for (int r = 0; r < 16; ++r) {
    const int i = rowBase + r;
    const float s1i = s1g[b * N + i];
    const float* arow = adjB + (size_t)i * N;
    float m = -3.0e38f, s = 0.f;
    for (int kb = 0; kb < N; kb += 32) {
      const int k = kb + lane;
      const float av  = arow[k];
      const float z   = s1i + s2b[k];
      const float lr  = (z > 0.f) ? z : ALPHA * z;
      const float val = (av > 0.f) ? lr : NEGC;
      const float mN  = fmaxf(m, val);
      s = s * __expf(m - mN) + __expf(val - mN);
      m = mN;
    }
    #pragma unroll
    for (int off = 16; off >= 1; off >>= 1) {
      const float mo = __shfl_xor(m, off, 32);
      const float so = __shfl_xor(s, off, 32);
      const float mN = fmaxf(m, mo);
      s = s * __expf(m - mN) + so * __expf(mo - mN);
      m = mN;
    }
    if (lane == 0) { smax[w][r] = m; ssum[w][r] = s; }
  }

  // ---- pass 2: WMMA over 32 k-chunks x 10 D-tiles ----
  v8f acc[10];
  #pragma unroll
  for (int dt = 0; dt < 10; ++dt) {
    #pragma unroll
    for (int r = 0; r < 8; ++r) acc[dt][r] = 0.f;
  }

  const int   myRow = rowBase + (lane & 15);
  const float s1i   = s1g[b * N + myRow];
  const float mi    = smax[w][lane & 15];
  const float* arowI = adjB + (size_t)myRow * N;
  const f16*   fragB = xW16 + (size_t)b * (32 * 10 * 512);
  const int    kHiSel = (lane >= 16) ? 8 : 0;

  for (int kc = 0; kc < 32; ++kc) {
    const f16* fc = fragB + (size_t)kc * (10 * 512);
    if (kc < 31) __builtin_prefetch(fc + 10 * 512, 0, 0);

    v16h av;
    #pragma unroll
    for (int e = 0; e < 16; ++e) {
      const int kk = e + ((e >= 8) ? 8 : 0) + kHiSel;   // A layout K
      const int k  = (kc << 5) + kk;
      const float aval = arowI[k];
      const float z    = s1i + s2b[k];
      const float lr   = (z > 0.f) ? z : ALPHA * z;
      const float val  = (aval > 0.f) ? lr : NEGC;
      av[e] = (_Float16)__expf(val - mi);               // unnormalized, in [0,1]
    }

    #pragma unroll
    for (int dt = 0; dt < 10; ++dt) {
      const v16h bv = *reinterpret_cast<const v16h*>(fc + ((size_t)(dt * 32 + lane) << 4));
      acc[dt] = WMMA(av, bv, acc[dt]);
    }
  }

  // ---- epilogue: normalize, ReLU, scatter into fragment buffers ----
  const int half = lane >> 4, laneLo = lane & 15;
  #pragma unroll
  for (int dt = 0; dt < 10; ++dt) {
    #pragma unroll
    for (int r = 0; r < 8; ++r) {
      const int M = r + 8 * half;
      float v = acc[dt][r] / ssum[w][M];
      v = (v > 0.f) ? v : 0.f;
      const int dcol = dt * 16 + laneLo;
      const int gRow = b * N + rowBase + M;
      const int rowTile = gRow >> 4;
      if (axwA) {  // layer-1 GEMM input (cols 150..159 are the zero pad)
        int laneHi, e; a_slot(dcol & 31, laneHi, e);
        axwA[(((size_t)rowTile * 5 + (dcol >> 5)) * 32 + (M + 16 * laneHi)) * 16 + e] = (f16)v;
      }
      if (dcol < D) {  // h = AxW + x at concat offset fBase (pads stay 0)
        const float hv = v + x[(size_t)gRow * IN + xBase + dcol];
        const int fp = fBase + dcol;
        int laneHi, e; a_slot(fp & 31, laneHi, e);
        h16[(((size_t)rowTile * 30 + (fp >> 5)) * 32 + (M + 16 * laneHi)) * 16 + e] = (f16)hv;
      }
    }
  }
}

// ---------------------------------------------------------------------------
// k_out_wmma: out = h @ Wout^T + bout.  One wave per 16-row tile,
// 30 K-chunks x 20 output tiles (acc = 160 VGPRs, fine for wave32).
// ---------------------------------------------------------------------------
__global__ void __launch_bounds__(128) k_out_wmma(
    const f16*  __restrict__ h16,    // [RT][30][32][16]
    const f16*  __restrict__ wofrag, // [30][20][32][16]
    const float* __restrict__ bout,  // [IN]
    float* __restrict__ out) {       // [BN][IN]
  const int w = threadIdx.x >> 5;
  const int lane = threadIdx.x & 31;
  const int rowTile = blockIdx.x * 4 + w;

  v8f acc[20];
  #pragma unroll
  for (int ot = 0; ot < 20; ++ot) {
    #pragma unroll
    for (int r = 0; r < 8; ++r) acc[ot][r] = 0.f;
  }

  for (int kc = 0; kc < 30; ++kc) {
    const v16h av = *reinterpret_cast<const v16h*>(
        h16 + (((size_t)rowTile * 30 + kc) * 32 + lane) * 16);
    const f16* wc = wofrag + (size_t)kc * (20 * 512);
    if (kc < 29) __builtin_prefetch(wc + 20 * 512, 0, 0);
    #pragma unroll
    for (int ot = 0; ot < 20; ++ot) {
      const v16h bv = *reinterpret_cast<const v16h*>(wc + ((size_t)(ot * 32 + lane) << 4));
      acc[ot] = WMMA(av, bv, acc[ot]);
    }
  }

  const int half = lane >> 4, laneLo = lane & 15;
  #pragma unroll
  for (int ot = 0; ot < 20; ++ot) {
    const int o = ot * 16 + laneLo;
    if (o < IN) {
      const float bo = bout[o];
      #pragma unroll
      for (int r = 0; r < 8; ++r) {
        const int gRow = rowTile * 16 + r + 8 * half;
        out[(size_t)gRow * IN + o] = acc[ot][r] + bo;
      }
    }
  }
}

// ---------------------------------------------------------------------------
// Launch.  Workspace (~27.7 MB), all f16 fragment buffers 256B-aligned:
//   xA16   [RT][10][512]  5.24 MB   x in A-layout (padded 320)
//   axwA16 [RT][ 5][512]  2.62 MB   AxW0 in A-layout (padded 160)
//   h16    [RT][30][512] 15.73 MB   concat h in A-layout (3 x 320)
//   xW16   [BB][32][10][512] 2.62MB xW in B-layout (per head-layer, reused)
//   w0frag 3x[10][10][512] 0.31 MB, w1frag 3x[15][10][512] 0.46 MB,
//   wofrag [30][20][512]  0.61 MB, s1/s2 64 KB
// ---------------------------------------------------------------------------
extern "C" void kernel_launch(void* const* d_in, const int* in_sizes, int n_in,
                              void* d_out, int out_size, void* d_ws, size_t ws_size,
                              hipStream_t stream) {
  (void)in_sizes; (void)n_in; (void)out_size; (void)ws_size;

  const float* x    = (const float*)d_in[0];
  const float* adj  = (const float*)d_in[1];
  const float* W0   = (const float*)d_in[2];
  const float* b0   = (const float*)d_in[3];
  const float* W1   = (const float*)d_in[4];
  const float* b1   = (const float*)d_in[5];
  const float* a    = (const float*)d_in[6];
  const float* Wout = (const float*)d_in[7];
  const float* bout = (const float*)d_in[8];
  float* out = (float*)d_out;

  char* ws = (char*)d_ws;
  size_t off = 0;
  auto take = [&](size_t bytes) {
    void* p = ws + off;
    off = (off + bytes + 255) & ~(size_t)255;
    return p;
  };
  f16* xA16   = (f16*)take((size_t)RT * 10 * 512 * sizeof(f16));
  f16* axwA16 = (f16*)take((size_t)RT * 5 * 512 * sizeof(f16));
  f16* h16    = (f16*)take((size_t)RT * 30 * 512 * sizeof(f16));
  f16* xW16   = (f16*)take((size_t)BB * 32 * 10 * 512 * sizeof(f16));
  f16* w0frag = (f16*)take((size_t)HEADS * 10 * 10 * 512 * sizeof(f16));
  f16* w1frag = (f16*)take((size_t)HEADS * 15 * 10 * 512 * sizeof(f16));
  f16* wofrag = (f16*)take((size_t)30 * 20 * 512 * sizeof(f16));
  float* s1   = (float*)take((size_t)BN * sizeof(float));
  float* s2   = (float*)take((size_t)BN * sizeof(float));

  const float* a1 = a;
  const float* a2 = a + D;

  // ---- prep: zero h pads, build x / weight fragments ----
  {
    const int nH = (RT * 30 * 512) / 2;  // halves -> u32 count
    k_zero_u32<<<(nH + 255) / 256, 256, 0, stream>>>((uint32_t*)h16, nH);
  }
  k_prep_x<<<BN, INP, 0, stream>>>(x, xA16);
  for (int j = 0; j < HEADS; ++j) {
    k_prep_w<<<(10 * 10 * 512 + 255) / 256, 256, 0, stream>>>(
        W0 + (size_t)j * D * IN, IN, 0, D, 10, 10, IN, IN,
        w0frag + (size_t)j * 10 * 10 * 512);
    k_prep_w<<<(15 * 10 * 512 + 255) / 256, 256, 0, stream>>>(
        W1 + (size_t)j * D * F1, F1, 0, D, 10, 15, IN, F1,
        w1frag + (size_t)j * 15 * 10 * 512);
    k_prep_w<<<(10 * 20 * 512 + 255) / 256, 256, 0, stream>>>(
        Wout, IN * HEADS, j * IN, IN, 20, 10, IN, IN,
        wofrag + (size_t)j * 10 * 20 * 512);
  }

  // ---- per-head pipeline ----
  for (int j = 0; j < HEADS; ++j) {
    // layer 0
    k_xw_wmma<<<RT / 4, 128, 0, stream>>>(
        xA16, nullptr, w0frag + (size_t)j * 10 * 10 * 512,
        b0 + j * D, a1, a2, 10, xW16, s1, s2);
    k_att<<<RT / 4, 128, 0, stream>>>(adj, s1, s2, xW16, x, h16, axwA16,
                                      j * INP + 0, 0);
    // layer 1 (outputs = [x | AxW0])
    k_xw_wmma<<<RT / 4, 128, 0, stream>>>(
        xA16, axwA16, w1frag + (size_t)j * 15 * 10 * 512,
        b1 + j * D, a1, a2, 15, xW16, s1, s2);
    k_att<<<RT / 4, 128, 0, stream>>>(adj, s1, s2, xW16, x, h16, nullptr,
                                      j * INP + D, D);
  }

  // ---- output projection ----
  k_out_wmma<<<RT / 4, 128, 0, stream>>>(h16, wofrag, bout, out);
}
